// Olmo3_5HybridGatedDeltaNet_35150012350724
// MI455X (gfx1250) — matrix operand order repaired and checked by reading, not compiled, not run on hardware
//
#include <hip/hip_runtime.h>

// ---------------------------------------------------------------------------
// Olmo3.5 Hybrid GatedDeltaNet for MI455X (gfx1250, wave32, WMMA)
// Large GEMMs: bf16 WMMA (v_wmma_f32_16x16x32_bf16), fp32 accumulate,
//              double-buffered LDS fed by global_load_async_to_lds_b128.
// Recurrence: register-resident state, LDS-staged chunks (async copies),
//             half-wave shuffle reductions.
// ---------------------------------------------------------------------------

#define T_TOK     2048
#define H_DIM     2048
#define NUM_HEADS 32
#define NUM_KV    16
#define DK        128
#define DV        128
#define KEY_DIM   2048   // NUM_KV * DK
#define VAL_DIM   4096   // NUM_HEADS * DV
#define MIX_DIM   8192   // 2*KEY_DIM + VAL_DIM

typedef __attribute__((ext_vector_type(16))) __bf16 v16bf;
typedef __attribute__((ext_vector_type(8)))  float  v8f;

union FragBF { uint4 u4[2]; v16bf v; };

__device__ __forceinline__ unsigned short f2bf(float f) {
    unsigned u = __float_as_uint(f);
    u = u + 0x7FFFu + ((u >> 16) & 1u);       // round-to-nearest-even
    return (unsigned short)(u >> 16);
}

__device__ __forceinline__ float sigf(float x) {
    return 1.0f / (1.0f + __expf(-x));
}

// CDNA5 async global->LDS DMA (tracked by ASYNCcnt). lds_off = low 32 bits of
// the generic pointer (LDS aperture offset). 16B per issue.
__device__ __forceinline__ void async_b128(unsigned lds_off, const void* gaddr) {
    asm volatile("global_load_async_to_lds_b128 %0, %1, off"
                 :: "v"(lds_off), "v"(gaddr) : "memory");
}
__device__ __forceinline__ unsigned lds_of(const void* p) {
    return (unsigned)(size_t)p;
}
__device__ __forceinline__ void wait_async0() {
#if __has_builtin(__builtin_amdgcn_s_wait_asynccnt)
    __builtin_amdgcn_s_wait_asynccnt(0);
#else
    asm volatile("s_wait_asynccnt 0x0" ::: "memory");
#endif
}

// ---------------------------------------------------------------------------
// fp32 -> bf16 bit conversion
// ---------------------------------------------------------------------------
__global__ void cvt_bf16(const float* __restrict__ in,
                         unsigned short* __restrict__ out, int n) {
    int i = blockIdx.x * 256 + threadIdx.x;
    if (i < n) out[i] = f2bf(in[i]);
}

// ---------------------------------------------------------------------------
// bf16 WMMA GEMM: C[M,N] (f32, pitch ldc) = A[M,K](bf16) @ B[K,N](bf16)
// Block: 256 threads = 8 waves. Tile: BM=128 x BN=64, BK=32.
// Double-buffered LDS; A tile streamed by async DMA; one barrier / K-step.
// ---------------------------------------------------------------------------
#define BM 128
#define BN 64
#define BK 32

__global__ __launch_bounds__(256)
void wmma_gemm_bf16(const unsigned short* __restrict__ A,
                    const unsigned short* __restrict__ B,
                    float* __restrict__ C,
                    int M, int N, int K, int lda, int ldb, int ldc) {
    __shared__ __align__(16) unsigned short As[2][BM * BK];   // [row][k]
    __shared__ __align__(16) unsigned short Bs[2][BN * BK];   // transposed [n][k]

    const int tid  = threadIdx.x;
    const int wave = tid >> 5;
    const int lane = tid & 31;
    const int lr   = lane & 15;       // M (A frag) / N (B,C frags) within 16
    const int hv   = lane >> 4;       // half-wave select
    const int row0 = blockIdx.x * BM;
    const int col0 = blockIdx.y * BN;

    // B staging (reg transpose): this thread owns k=kk, cols col0+ng*8..+7
    const int kk = tid & 31;
    const int ng = tid >> 5;

    v8f acc[4] = {};
    union BReg { uint4 u; unsigned short s[8]; };

    // ---- prologue: stage tile k0=0 into buffer 0 ----
    {
        int ci = tid << 1;
        #pragma unroll
        for (int u = 0; u < 2; ++u, ++ci) {
            int row = ci >> 2, ko = (ci & 3) << 3;
            async_b128(lds_of(&As[0][(row << 5) + ko]),
                       &A[(size_t)(row0 + row) * lda + ko]);
        }
        BReg br;
        br.u = *(const uint4*)&B[(size_t)kk * ldb + col0 + (ng << 3)];
        #pragma unroll
        for (int i = 0; i < 8; ++i)
            Bs[0][((ng << 3) + i) * BK + kk] = br.s[i];
    }
    wait_async0();
    __syncthreads();

    for (int k0 = 0; k0 < K; k0 += BK) {
        const int  cur     = (k0 >> 5) & 1;
        const int  nxt     = cur ^ 1;
        const bool hasNext = (k0 + BK) < K;

        // ---- issue next tile's loads before computing (latency overlap) ----
        BReg brn;
        if (hasNext) {
            int ci = tid << 1;
            #pragma unroll
            for (int u = 0; u < 2; ++u, ++ci) {
                int row = ci >> 2, ko = (ci & 3) << 3;
                async_b128(lds_of(&As[nxt][(row << 5) + ko]),
                           &A[(size_t)(row0 + row) * lda + (k0 + BK) + ko]);
            }
            brn.u = *(const uint4*)&B[(size_t)(k0 + BK + kk) * ldb + col0 + (ng << 3)];
        }

        // ---- fragments from current buffer ----
        // A (ISA 16-bit 16x32): lane lr=M, hv selects K bases {0,8};
        // element runs kb..kb+7 and kb+16..kb+23 -> two 16B LDS loads.
        FragBF af;
        const unsigned short* ap = &As[cur][((wave << 4) + lr) * BK + (hv << 3)];
        af.u4[0] = *(const uint4*)(ap);
        af.u4[1] = *(const uint4*)(ap + 16);
        // B: lane lr=N, hv selects K bases {0,8}; runs kb..kb+7 and kb+16..kb+23 (matches A).
        FragBF bfr[4];
        #pragma unroll
        for (int j = 0; j < 4; ++j) {
            const unsigned short* bp = &Bs[cur][((j << 4) + lr) * BK + (hv << 3)];
            bfr[j].u4[0] = *(const uint4*)(bp);
            bfr[j].u4[1] = *(const uint4*)(bp + 16);
        }
        #pragma unroll
        for (int j = 0; j < 4; ++j)
            acc[j] = __builtin_amdgcn_wmma_f32_16x16x32_bf16(
                false, af.v, false, bfr[j].v, (short)0, acc[j], false, false);

        if (hasNext) {
            #pragma unroll
            for (int i = 0; i < 8; ++i)
                Bs[nxt][((ng << 3) + i) * BK + kk] = brn.s[i];
            wait_async0();
            __syncthreads();
        }
    }

    // ---- store: C/D layout lane=(N), VGPR r -> M = r + 8*hv ----
    const int mbase = row0 + (wave << 4) + (hv << 3);
    #pragma unroll
    for (int j = 0; j < 4; ++j) {
        #pragma unroll
        for (int r = 0; r < 8; ++r)
            C[(size_t)(mbase + r) * ldc + col0 + (j << 4) + lr] = acc[j][r];
    }
}

// ---------------------------------------------------------------------------
// g = -exp(A_log)*softplus(h@Wa + dt_bias);  beta = sigmoid(h@Wb)
// ---------------------------------------------------------------------------
__global__ void proj_ab(const float* __restrict__ h,
                        const float* __restrict__ Wa,
                        const float* __restrict__ Wb,
                        const float* __restrict__ dt_bias,
                        const float* __restrict__ A_log,
                        float* __restrict__ gvec,
                        float* __restrict__ betav) {
    int n = blockIdx.x * 256 + threadIdx.x;
    if (n >= T_TOK * NUM_HEADS) return;
    int t  = n >> 5;
    int hd = n & 31;
    const float* hr = h + (size_t)t * H_DIM;
    float a = 0.f, b = 0.f;
    for (int i = 0; i < H_DIM; ++i) {
        float x = hr[i];
        a += x * Wa[i * NUM_HEADS + hd];
        b += x * Wb[i * NUM_HEADS + hd];
    }
    float x  = a + dt_bias[hd];
    float sp = (x <= 20.f) ? logf(1.0f + __expf(x)) : x;
    gvec[n]  = -__expf(A_log[hd]) * sp;
    betav[n] = sigf(b);
}

// ---------------------------------------------------------------------------
// Causal depthwise conv (K=4) + SiLU over concatenated qkv.
// ---------------------------------------------------------------------------
__global__ void conv_silu(const float* __restrict__ pre,
                          const float* __restrict__ cq,
                          const float* __restrict__ ck,
                          const float* __restrict__ cv,
                          float* __restrict__ out) {
    int n = blockIdx.x * 256 + threadIdx.x;
    if (n >= T_TOK * MIX_DIM) return;
    int t = n >> 13;
    int c = n & (MIX_DIM - 1);
    const float* w;
    if (c < KEY_DIM)            w = cq + (size_t)c * 4;
    else if (c < 2 * KEY_DIM)   w = ck + (size_t)(c - KEY_DIM) * 4;
    else                        w = cv + (size_t)(c - 2 * KEY_DIM) * 4;
    float acc = 0.f;
    #pragma unroll
    for (int j = 0; j < 4; ++j) {
        int tt = t - 3 + j;
        if (tt >= 0) acc += w[j] * pre[(size_t)tt * MIX_DIM + c];
    }
    out[n] = acc * sigf(acc);
}

// ---------------------------------------------------------------------------
// L2-normalize q and k rows (per kv-head, 128 elems); q also * DK^-0.5.
// ---------------------------------------------------------------------------
__global__ __launch_bounds__(256)
void l2norm_qk(const float* __restrict__ mixed,
               float* __restrict__ qn, float* __restrict__ kn) {
    int wid  = (blockIdx.x * 256 + threadIdx.x) >> 5;   // 2*T*NUM_KV waves
    int lane = threadIdx.x & 31;
    int which = (wid >= T_TOK * NUM_KV) ? 1 : 0;        // 0=q, 1=k
    int rr    = wid - which * (T_TOK * NUM_KV);
    int t     = rr >> 4;
    int kvh   = rr & 15;
    const float* src = mixed + (size_t)t * MIX_DIM + which * KEY_DIM
                       + kvh * DK + lane * 4;
    float4 x = *(const float4*)src;
    float ss = x.x * x.x + x.y * x.y + x.z * x.z + x.w * x.w;
    #pragma unroll
    for (int m = 16; m >= 1; m >>= 1) ss += __shfl_xor(ss, m, 32);
    float s = rsqrtf(ss + 1e-6f);
    if (which == 0) s *= 0.08838834764831845f;          // DK^-0.5
    float* dst = (which ? kn : qn) + ((size_t)t * NUM_KV + kvh) * DK + lane * 4;
    float4 y = { x.x * s, x.y * s, x.z * s, x.w * s };
    *(float4*)dst = y;
}

// ---------------------------------------------------------------------------
// Gated delta-rule recurrence. 256 blocks = 32 heads x 8 v-column blocks.
// Thread owns 8 state rows of one v-column (registers). Half-wave (16 lanes)
// covers the 128-row K dim; shfl_xor reductions. 32 timesteps of k/q/v/g/beta
// are staged in LDS per chunk (k/q via async DMA, v/g/beta via direct stores).
// ---------------------------------------------------------------------------
#define SCH 32

__global__ __launch_bounds__(256)
void gdn_scan(const float* __restrict__ qn, const float* __restrict__ kn,
              const float* __restrict__ mixed,
              const float* __restrict__ gvec, const float* __restrict__ betav,
              float* __restrict__ o) {
    __shared__ __align__(16) float kbuf[SCH][DK];   // 16 KB
    __shared__ __align__(16) float qbuf[SCH][DK];   // 16 KB
    __shared__ float vbuf[SCH][16];                 //  2 KB
    __shared__ float gbuf[SCH];
    __shared__ float bbuf[SCH];

    const int head = blockIdx.x >> 3;
    const int vb   = blockIdx.x & 7;
    const int kvh  = head >> 1;                  // GQA: repeat-interleave by 2
    const int tid  = threadIdx.x;
    const int wave = tid >> 5;
    const int lane = tid & 31;
    const int hv   = lane >> 4;
    const int lr   = lane & 15;
    const int colL = wave * 2 + hv;              // local v-column (0..15)
    const int col  = vb * 16 + colL;             // global v-column in head
    const int row0 = lr * 8;                     // 8 K-rows owned by this lane

    float S[8];
    #pragma unroll
    for (int i = 0; i < 8; ++i) S[i] = 0.f;

    // staging coords: thread copies 16 floats of step si for k and q
    const int si = tid >> 3;                     // 0..31
    const int so = (tid & 7) * 16;               // float offset in row
    const int vi = tid >> 4;                     // 0..15 (+16 second half)
    const int vc = tid & 15;

    for (int t0 = 0; t0 < T_TOK; t0 += SCH) {
        // ---- stage chunk: k/q via async DMA, v/g/beta via direct stores ----
        {
            const float* kg = kn + ((size_t)(t0 + si) * NUM_KV + kvh) * DK + so;
            const float* qg = qn + ((size_t)(t0 + si) * NUM_KV + kvh) * DK + so;
            #pragma unroll
            for (int u = 0; u < 4; ++u) {
                async_b128(lds_of(&kbuf[si][so + u * 4]), kg + u * 4);
                async_b128(lds_of(&qbuf[si][so + u * 4]), qg + u * 4);
            }
            const size_t vgoff = 2 * KEY_DIM + head * DV + vb * 16 + vc;
            vbuf[vi][vc]      = mixed[(size_t)(t0 + vi)      * MIX_DIM + vgoff];
            vbuf[vi + 16][vc] = mixed[(size_t)(t0 + vi + 16) * MIX_DIM + vgoff];
            if (tid < SCH) {
                gbuf[tid] = gvec[(t0 + tid) * NUM_HEADS + head];
                bbuf[tid] = betav[(t0 + tid) * NUM_HEADS + head];
            }
        }
        wait_async0();
        __syncthreads();

        // ---- 32 recurrence steps from LDS ----
        for (int i = 0; i < SCH; ++i) {
            float eg = __expf(gbuf[i]);
            float bt = bbuf[i];
            float4 k0 = *(const float4*)&kbuf[i][row0];
            float4 k1 = *(const float4*)&kbuf[i][row0 + 4];
            float4 q0 = *(const float4*)&qbuf[i][row0];
            float4 q1 = *(const float4*)&qbuf[i][row0 + 4];
            float vv = vbuf[i][colL];
            float kk[8] = {k0.x, k0.y, k0.z, k0.w, k1.x, k1.y, k1.z, k1.w};
            float qq[8] = {q0.x, q0.y, q0.z, q0.w, q1.x, q1.y, q1.z, q1.w};

            float kvp = 0.f;
            #pragma unroll
            for (int r = 0; r < 8; ++r) { S[r] *= eg; kvp += kk[r] * S[r]; }
            #pragma unroll
            for (int m = 8; m >= 1; m >>= 1) kvp += __shfl_xor(kvp, m, 32);

            float dv = bt * (vv - kvp);

            float op = 0.f;
            #pragma unroll
            for (int r = 0; r < 8; ++r) { S[r] += kk[r] * dv; op += qq[r] * S[r]; }
            #pragma unroll
            for (int m = 8; m >= 1; m >>= 1) op += __shfl_xor(op, m, 32);

            if (lr == 0)
                o[((size_t)(t0 + i) * NUM_HEADS + head) * DV + col] = op;
        }
        __syncthreads();
    }
}

// ---------------------------------------------------------------------------
// Gated RMSNorm (eps=1e-5) * o_norm_weight * silu(gate); emit bf16.
// ---------------------------------------------------------------------------
__global__ __launch_bounds__(256)
void onorm_gate(const float* __restrict__ o, const float* __restrict__ gate,
                const float* __restrict__ w, unsigned short* __restrict__ onbf) {
    int wid  = (blockIdx.x * 256 + threadIdx.x) >> 5;   // T*NUM_HEADS waves
    int lane = threadIdx.x & 31;
    int t  = wid >> 5;
    int hd = wid & 31;
    const float* op = o + ((size_t)t * NUM_HEADS + hd) * DV + lane * 4;
    float4 x = *(const float4*)op;
    float ss = x.x * x.x + x.y * x.y + x.z * x.z + x.w * x.w;
    #pragma unroll
    for (int m = 16; m >= 1; m >>= 1) ss += __shfl_xor(ss, m, 32);
    float r = rsqrtf(ss * (1.0f / 128.0f) + 1e-5f);
    const float* gp = gate + (size_t)t * VAL_DIM + hd * DV + lane * 4;
    float4 gg = *(const float4*)gp;
    const float* wp = w + lane * 4;
    float y0 = x.x * r * wp[0] * (gg.x * sigf(gg.x));
    float y1 = x.y * r * wp[1] * (gg.y * sigf(gg.y));
    float y2 = x.z * r * wp[2] * (gg.z * sigf(gg.z));
    float y3 = x.w * r * wp[3] * (gg.w * sigf(gg.w));
    ushort4 pk = { f2bf(y0), f2bf(y1), f2bf(y2), f2bf(y3) };
    *(ushort4*)&onbf[(size_t)t * VAL_DIM + hd * DV + lane * 4] = pk;
}

// ---------------------------------------------------------------------------
// Host orchestration
// ---------------------------------------------------------------------------
extern "C" void kernel_launch(void* const* d_in, const int* in_sizes, int n_in,
                              void* d_out, int out_size, void* d_ws, size_t ws_size,
                              hipStream_t stream) {
    const float* h       = (const float*)d_in[0];
    const float* Wq      = (const float*)d_in[1];
    const float* Wk      = (const float*)d_in[2];
    const float* Wv      = (const float*)d_in[3];
    const float* Wa      = (const float*)d_in[4];
    const float* Wb      = (const float*)d_in[5];
    const float* Wg      = (const float*)d_in[6];
    const float* conv_q  = (const float*)d_in[7];
    const float* conv_k  = (const float*)d_in[8];
    const float* conv_v  = (const float*)d_in[9];
    const float* dt_bias = (const float*)d_in[10];
    const float* A_log   = (const float*)d_in[11];
    const float* o_normw = (const float*)d_in[12];
    const float* Wo      = (const float*)d_in[13];
    float*       out     = (float*)d_out;

    char* ws = (char*)d_ws;
    const size_t MB = 1ull << 20;
    unsigned short* h_bf  = (unsigned short*)(ws + 0);          //  8 MiB
    unsigned short* wq_bf = (unsigned short*)(ws + 8 * MB);     //  8 MiB
    unsigned short* wk_bf = (unsigned short*)(ws + 16 * MB);    //  8 MiB
    unsigned short* wv_bf = (unsigned short*)(ws + 24 * MB);    // 16 MiB
    unsigned short* wg_bf = (unsigned short*)(ws + 40 * MB);    // 16 MiB
    unsigned short* wo_bf = (unsigned short*)(ws + 56 * MB);    // 16 MiB
    float*          pre   = (float*)(ws + 72 * MB);             // 64 MiB
    float*          mixed = (float*)(ws + 136 * MB);            // 64 MiB
    float*          gate  = (float*)(ws + 200 * MB);            // 32 MiB
    float*          gvec  = (float*)(ws + 232 * MB);            // 256 KiB
    float*          betav = (float*)(ws + 232 * MB + 256 * 1024);
    // Overlays (lifetimes disjoint, stream-ordered):
    float* qn    = pre;                                         // 16 MiB
    float* kn    = (float*)((char*)pre + 16 * MB);              // 16 MiB
    float* ocore = (float*)((char*)pre + 32 * MB);              // 32 MiB
    unsigned short* onbf = wq_bf;                               // 16 MiB (wq+wk)

    // 1. fp32 -> bf16 conversions
    auto cvt = [&](const float* src, unsigned short* dst, int n) {
        cvt_bf16<<<(n + 255) / 256, 256, 0, stream>>>(src, dst, n);
    };
    cvt(h,  h_bf,  T_TOK * H_DIM);
    cvt(Wq, wq_bf, H_DIM * KEY_DIM);
    cvt(Wk, wk_bf, H_DIM * KEY_DIM);
    cvt(Wv, wv_bf, H_DIM * VAL_DIM);
    cvt(Wg, wg_bf, H_DIM * VAL_DIM);
    cvt(Wo, wo_bf, VAL_DIM * H_DIM);

    // 2. Projections (WMMA): q|k|v written into fused pre[T, 8192]
    dim3 blk(256);
    dim3 gN2(T_TOK / BM, KEY_DIM / BN);
    dim3 gN4(T_TOK / BM, VAL_DIM / BN);
    wmma_gemm_bf16<<<gN2, blk, 0, stream>>>(h_bf, wq_bf, pre + 0,
        T_TOK, KEY_DIM, H_DIM, H_DIM, KEY_DIM, MIX_DIM);
    wmma_gemm_bf16<<<gN2, blk, 0, stream>>>(h_bf, wk_bf, pre + KEY_DIM,
        T_TOK, KEY_DIM, H_DIM, H_DIM, KEY_DIM, MIX_DIM);
    wmma_gemm_bf16<<<gN4, blk, 0, stream>>>(h_bf, wv_bf, pre + 2 * KEY_DIM,
        T_TOK, VAL_DIM, H_DIM, H_DIM, VAL_DIM, MIX_DIM);
    wmma_gemm_bf16<<<gN4, blk, 0, stream>>>(h_bf, wg_bf, gate,
        T_TOK, VAL_DIM, H_DIM, H_DIM, VAL_DIM, VAL_DIM);

    // 3. Gating scalars g, beta
    proj_ab<<<(T_TOK * NUM_HEADS + 255) / 256, 256, 0, stream>>>(
        h, Wa, Wb, dt_bias, A_log, gvec, betav);

    // 4. Causal depthwise conv + SiLU
    conv_silu<<<(T_TOK * MIX_DIM) / 256, 256, 0, stream>>>(
        pre, conv_q, conv_k, conv_v, mixed);

    // 5. L2 norm of q/k per kv-head (q also scaled by DK^-0.5)
    l2norm_qk<<<(2 * T_TOK * NUM_KV * 32) / 256, 256, 0, stream>>>(mixed, qn, kn);

    // 6. Gated delta-rule recurrence (LDS-staged chunks)
    gdn_scan<<<NUM_HEADS * 8, 256, 0, stream>>>(qn, kn, mixed, gvec, betav, ocore);

    // 7. Gated RMSNorm + silu(gate), emit bf16
    onorm_gate<<<(T_TOK * NUM_HEADS * 32) / 256, 256, 0, stream>>>(
        ocore, gate, o_normw, onbf);

    // 8. Output projection (WMMA) -> d_out
    dim3 gO(T_TOK / BM, H_DIM / BN);
    wmma_gemm_bf16<<<gO, blk, 0, stream>>>(onbf, wo_bf, out,
        T_TOK, H_DIM, VAL_DIM, VAL_DIM, H_DIM, H_DIM);
}
